// RVQVAE_67886253081219
// MI455X (gfx1250) — matrix operand order repaired and checked by reading, not compiled, not run on hardware
//
#include <hip/hip_runtime.h>
#include <hip/hip_bf16.h>
#include <stdint.h>

// ---------------------------------------------------------------------------
// Types for CDNA5 WMMA (gfx1250, wave32): D(f32 16x16) = A(bf16 16x32) x B(bf16 32x16) + C
// ---------------------------------------------------------------------------
typedef __bf16  bf16_t;
typedef bf16_t  v16bf __attribute__((ext_vector_type(16)));
typedef float   v8f   __attribute__((ext_vector_type(8)));

__device__ __forceinline__ unsigned short f2bf(float f) {
    unsigned int u = __builtin_bit_cast(unsigned int, f);
    unsigned int r = u + 0x7FFFu + ((u >> 16) & 1u);   // round-to-nearest-even
    return (unsigned short)(r >> 16);
}
__device__ __forceinline__ float bf2f(unsigned short h) {
    unsigned int u = ((unsigned int)h) << 16;
    return __builtin_bit_cast(float, u);
}

#define F_RELU_IN  1
#define F_RELU_OUT 2
#define F_UP2      4

#define BM 128
#define BN 128
#define BK 32
#define LDA 40   // shorts/row: 32 data + 8 pad; row pitch 80B (16B aligned)
#define LDB 40

// ---------------------------------------------------------------------------
// Weight pre-pack: fp32 (Co, Kg) -> bf16 (CoP, KgP), zero padded.
// Makes the conv A-tile a guard-free contiguous copy (async-to-LDS friendly).
// ---------------------------------------------------------------------------
__global__ void pack_w(const float* __restrict__ w, unsigned short* __restrict__ out,
                       int Co, int Kg, int CoP, int KgP) {
    const int idx = blockIdx.x * blockDim.x + threadIdx.x;
    const int total = CoP * KgP;
    if (idx >= total) return;
    const int k = idx % KgP;
    const int o = idx / KgP;
    out[idx] = (o < Co && k < Kg) ? f2bf(w[(size_t)o * Kg + k]) : (unsigned short)0;
}

// ---------------------------------------------------------------------------
// Generic conv1d as implicit GEMM with WMMA bf16, double-buffered pipeline:
//   while tile s is multiplied (8x v_wmma), tile s+1's weights are already in
//   flight via global_load_async_to_lds_b128 and its activation patches are
//   being gathered into registers; one barrier per k-step.
// ---------------------------------------------------------------------------
struct ConvParams {
    const unsigned short* X;     // (N, Ci, Tin) bf16
    const unsigned short* Wb;    // packed bf16 (CoP, KgP)
    const float* Bias;           // (Co,)
    const unsigned short* Res;   // optional residual, (N, Co, Tout) bf16
    unsigned short* Ybf;         // bf16 output (or null)
    float* Yf;                   // fp32 output (or null)
    int Ci, Tin, Co, Tout, Kw, stride, dil, pad, Kg, KgP, flags;
};

__global__ __launch_bounds__(256) void conv1d_wmma(ConvParams p) {
    __shared__ __align__(16) unsigned short sA[2][BM * LDA];
    __shared__ __align__(16) unsigned short sB[2][BN * LDB];

    const int tid  = threadIdx.x;
    const int lane = tid & 31;
    const int wv   = tid >> 5;          // wave 0..7
    const int wm   = (wv & 3) * 32;     // wave M offset in block tile
    const int wn   = (wv >> 2) * 64;    // wave N offset in block tile
    const int half = lane >> 4;
    const int l16  = lane & 15;

    const int b  = blockIdx.z;
    const int m0 = blockIdx.y * BM;     // grid.y = CoP/128, so m0+127 < CoP always
    const int n0 = blockIdx.x * BN;

    const bool relu_in  = (p.flags & F_RELU_IN)  != 0;
    const bool relu_out = (p.flags & F_RELU_OUT) != 0;
    const bool up2      = (p.flags & F_UP2)      != 0;
    const int  TinL     = up2 ? 2 * p.Tin : p.Tin;

    v8f acc[2][4];
    #pragma unroll
    for (int mt = 0; mt < 2; mt++)
        #pragma unroll
        for (int nt = 0; nt < 4; nt++)
            #pragma unroll
            for (int e = 0; e < 8; e++) acc[mt][nt][e] = 0.0f;

    // ---- A async-copy mapping: 512 transfers of 16B; 2 per thread ----
    //   q = tid*2+e : row = q>>2 (0..127), seg = q&3 (16B segment within 64B row)
    const int q0    = tid * 2;
    const int arow0 = (q0)     >> 2, aseg0 = (q0)     & 3;
    const int arow1 = (q0 + 1) >> 2, aseg1 = (q0 + 1) & 3;
    const unsigned short* wg0 = p.Wb + (size_t)(m0 + arow0) * p.KgP + aseg0 * 8;
    const unsigned short* wg1 = p.Wb + (size_t)(m0 + arow1) * p.KgP + aseg1 * 8;

    // ---- B mapping: one output position per thread, 16 consecutive k ----
    const int nl = tid >> 1;            // 0..127
    const int kb = (tid & 1) * 16;      // 0 or 16
    const int t  = n0 + nl;
    const bool tok = (t < p.Tout);
    const int tb = t * p.stride - p.pad;
    const unsigned short* Xb = p.X + (size_t)b * p.Ci * p.Tin;

    // issue async weight copy of one 128x32 tile into LDS buffer `buf`
    auto stageA = [&](int kg0, int buf) {
        const unsigned int la0 = (unsigned int)(uintptr_t)&sA[buf][arow0 * LDA + aseg0 * 8];
        const unsigned int la1 = (unsigned int)(uintptr_t)&sA[buf][arow1 * LDA + aseg1 * 8];
        asm volatile("global_load_async_to_lds_b128 %0, %1, off"
                     :: "v"(la0), "v"(wg0 + kg0) : "memory");
        asm volatile("global_load_async_to_lds_b128 %0, %1, off"
                     :: "v"(la1), "v"(wg1 + kg0) : "memory");
    };
    // branchless clamped gather of this thread's 16 patch elements -> 8 dwords
    auto gatherB = [&](int kg0, unsigned int* pk) {
        const int kgs = kg0 + kb;
        int ci = kgs / p.Kw;
        int kw = kgs - ci * p.Kw;
        #pragma unroll
        for (int j = 0; j < 16; j++) {
            const int  kg = kgs + j;
            const int  ti = tb + kw * p.dil;
            const bool ok = tok && (kg < p.Kg) && (ti >= 0) && (ti < TinL);
            int tis = up2 ? (ti >> 1) : ti;
            tis = tis < 0 ? 0 : (tis >= p.Tin ? p.Tin - 1 : tis);
            const int cis = ci < p.Ci ? ci : p.Ci - 1;
            unsigned short v = Xb[(size_t)cis * p.Tin + tis];   // unconditional load
            if (!ok) v = 0;
            if (relu_in && (v & 0x8000u)) v = 0;
            if (j & 1) pk[j >> 1] |= ((unsigned int)v) << 16;
            else       pk[j >> 1]  = v;
            kw++;
            if (kw == p.Kw) { kw = 0; ci++; }
        }
    };
    auto storeB = [&](const unsigned int* pk, int buf) {
        *(uint4*)&sB[buf][nl * LDB + kb]     = *(const uint4*)&pk[0];
        *(uint4*)&sB[buf][nl * LDB + kb + 8] = *(const uint4*)&pk[4];
    };

    const int nk = (p.Kg + BK - 1) / BK;

    // prologue: stage tile 0 into buffer 0
    {
        unsigned int pk[8];
        stageA(0, 0);
        gatherB(0, pk);
        storeB(pk, 0);
        asm volatile("s_wait_asynccnt 0x0" ::: "memory");
        __syncthreads();
    }

    for (int s = 0; s < nk; s++) {
        const int cb = s & 1;
        const int nb = cb ^ 1;
        const bool have_next = (s + 1) < nk;

        unsigned int pk[8];
        if (have_next) {
            stageA((s + 1) * BK, nb);      // async weights for tile s+1
            gatherB((s + 1) * BK, pk);     // global u16 loads complete under the WMMAs
        }

        // ---- fragments from current buffer per documented CDNA5 VGPR layouts ----
        union U { unsigned int u[8]; v16bf v; } afr[2], bfr[4];
        #pragma unroll
        for (int mt = 0; mt < 2; mt++) {
            const int m = wm + mt * 16 + l16;
            #pragma unroll
            for (int r = 0; r < 8; r++) {
                const int k0 = 2 * r + 8 * half + ((r >= 4) ? 8 : 0);   // A 16x32 bf16
                afr[mt].u[r] = *(const unsigned int*)&sA[cb][m * LDA + k0];
            }
        }
        #pragma unroll
        for (int nt = 0; nt < 4; nt++) {
            const int n = wn + nt * 16 + l16;
            #pragma unroll
            for (int r = 0; r < 8; r++) {
                const int k0 = 2 * r + 8 * half + ((r >= 4) ? 8 : 0);   // B 32x16 bf16
                bfr[nt].u[r] = *(const unsigned int*)&sB[cb][n * LDB + k0];
            }
        }
        #pragma unroll
        for (int mt = 0; mt < 2; mt++)
            #pragma unroll
            for (int nt = 0; nt < 4; nt++)
                acc[mt][nt] = __builtin_amdgcn_wmma_f32_16x16x32_bf16(
                    false, afr[mt].v, false, bfr[nt].v,
                    (short)0, acc[mt][nt], false, false);

        if (have_next) storeB(pk, nb);     // next buffer's readers finished last barrier
        asm volatile("s_wait_asynccnt 0x0" ::: "memory");
        __syncthreads();
    }

    // ---- epilogue: bias, residual, relu, store ----
    #pragma unroll
    for (int mt = 0; mt < 2; mt++) {
        #pragma unroll
        for (int nt = 0; nt < 4; nt++) {
            const int tt = n0 + wn + nt * 16 + l16;
            if (tt >= p.Tout) continue;
            #pragma unroll
            for (int r = 0; r < 8; r++) {
                const int m = m0 + wm + mt * 16 + r + 8 * half;   // C/D: M = r + 8*half
                if (m >= p.Co) continue;
                float val = acc[mt][nt][r] + p.Bias[m];
                const size_t oidx = ((size_t)b * p.Co + m) * p.Tout + tt;
                if (p.Res) val += bf2f(p.Res[oidx]);
                if (relu_out) val = val > 0.0f ? val : 0.0f;
                if (p.Yf) p.Yf[oidx] = val;
                else      p.Ybf[oidx] = f2bf(val);
            }
        }
    }
}

// ---------------------------------------------------------------------------
// Input pack: x (N, T, C) fp32 -> Xp (N, C, T) bf16
// ---------------------------------------------------------------------------
__global__ void pack_input(const float* __restrict__ x, unsigned short* __restrict__ Xp,
                           int N, int T, int C) {
    const int idx = blockIdx.x * blockDim.x + threadIdx.x;
    const int total = N * C * T;
    if (idx >= total) return;
    const int t = idx % T;
    const int c = (idx / T) % C;
    const int n = idx / (T * C);
    Xp[idx] = f2bf(x[((size_t)n * T + t) * C + c]);
}

// ---------------------------------------------------------------------------
// LFQ stage 1: per sample s=(n,t): zp = z@wi+bi; commit; softmax entropy over
// the implicit 1024 sign-codebook. Deterministic tree reductions only.
// ---------------------------------------------------------------------------
__global__ __launch_bounds__(256) void lfq_stage1(
        const unsigned short* __restrict__ zenc,   // (N, 512, Tq) bf16
        const float* __restrict__ wi,              // (512, 10)
        const float* __restrict__ bi,              // (10,)
        float* __restrict__ zp_out,                // (S, 10)
        float* __restrict__ ent_out,               // (S,)
        float* __restrict__ commit_out,            // (S,)
        float* __restrict__ prob_out,              // (S, 1024)
        int Tq, int D) {
    __shared__ float red[256];
    __shared__ float zpS[10];
    const int s = blockIdx.x;
    const int n = s / Tq;
    const int t = s - n * Tq;
    const int tid = threadIdx.x;

    float part[10];
    #pragma unroll
    for (int j = 0; j < 10; j++) part[j] = 0.0f;
    for (int d = tid; d < D; d += 256) {
        const float zd = bf2f(zenc[((size_t)n * D + d) * Tq + t]);
        #pragma unroll
        for (int j = 0; j < 10; j++) part[j] += zd * wi[d * 10 + j];
    }
    for (int j = 0; j < 10; j++) {
        red[tid] = part[j];
        __syncthreads();
        for (int o = 128; o > 0; o >>= 1) {
            if (tid < o) red[tid] += red[tid + o];
            __syncthreads();
        }
        if (tid == 0) zpS[j] = red[0] + bi[j];
        __syncthreads();
    }
    if (tid < 10) zp_out[s * 10 + tid] = zpS[tid];
    if (tid == 0) {
        float c = 0.0f;
        #pragma unroll
        for (int j = 0; j < 10; j++) {
            const float q = zpS[j] > 0.0f ? 1.0f : -1.0f;
            const float d2 = zpS[j] - q;
            c += d2 * d2;
        }
        commit_out[s] = c;
    }
    __syncthreads();

    float lg[4];
    float lmax = -1e30f;
    #pragma unroll
    for (int r = 0; r < 4; r++) {
        const int c = tid + r * 256;
        float dot = 0.0f;
        #pragma unroll
        for (int j = 0; j < 10; j++)
            dot += ((c >> (9 - j)) & 1) ? zpS[j] : -zpS[j];
        lg[r] = 200.0f * dot;                       // 2*INV_TEMP
        lmax = fmaxf(lmax, lg[r]);
    }
    red[tid] = lmax; __syncthreads();
    for (int o = 128; o > 0; o >>= 1) { if (tid < o) red[tid] = fmaxf(red[tid], red[tid + o]); __syncthreads(); }
    lmax = red[0]; __syncthreads();

    float se = 0.0f, spl = 0.0f;
    float pr[4];
    #pragma unroll
    for (int r = 0; r < 4; r++) {
        pr[r] = __expf(lg[r] - lmax);
        se  += pr[r];
        spl += pr[r] * lg[r];
    }
    red[tid] = se; __syncthreads();
    for (int o = 128; o > 0; o >>= 1) { if (tid < o) red[tid] += red[tid + o]; __syncthreads(); }
    const float Z = red[0]; __syncthreads();
    red[tid] = spl; __syncthreads();
    for (int o = 128; o > 0; o >>= 1) { if (tid < o) red[tid] += red[tid + o]; __syncthreads(); }
    const float SPL = red[0]; __syncthreads();

    if (tid == 0) {
        const float lse = lmax + __logf(Z);
        ent_out[s] = lse - SPL / Z;                 // -sum p*logp
    }
    const float invZ = 1.0f / Z;
    #pragma unroll
    for (int r = 0; r < 4; r++)
        prob_out[(size_t)s * 1024 + tid + r * 256] = pr[r] * invZ;
}

__global__ void lfq_stage2(const float* __restrict__ prob, float* __restrict__ term, int S) {
    const int c = blockIdx.x * blockDim.x + threadIdx.x;
    if (c >= 1024) return;
    float s = 0.0f;
    for (int i = 0; i < S; i++) s += prob[(size_t)i * 1024 + c];
    const float avg = s / (float)S;
    term[c] = -avg * __logf(avg + 1e-9f);
}

__global__ __launch_bounds__(256) void lfq_stage3(
        const float* __restrict__ ent, const float* __restrict__ commit,
        const float* __restrict__ term, float* __restrict__ aux, int S) {
    __shared__ float red[256];
    const int tid = threadIdx.x;
    float a = 0.0f, b = 0.0f, c = 0.0f;
    for (int i = tid; i < S; i += 256) { a += ent[i]; b += commit[i]; }
    for (int i = tid; i < 1024; i += 256) c += term[i];
    float sums[3] = {a, b, c};
    float out[3];
    for (int k = 0; k < 3; k++) {
        red[tid] = sums[k]; __syncthreads();
        for (int o = 128; o > 0; o >>= 1) { if (tid < o) red[tid] += red[tid + o]; __syncthreads(); }
        out[k] = red[0]; __syncthreads();
    }
    if (tid == 0) {
        const float pse = out[0] / (float)S;
        const float cm  = out[1] / ((float)S * 10.0f);
        const float cbe = out[2];
        aux[0] = 0.1f * (pse - 1.0f * cbe) + 0.25f * cm;
    }
}

// zq = sign(zp) @ wo + bo, transposed to (N, 512, Tq) bf16 for the decoder
__global__ void lfq_out(const float* __restrict__ zp, const float* __restrict__ wo,
                        const float* __restrict__ bo, unsigned short* __restrict__ zq,
                        int Tq, int E, int total) {
    const int idx = blockIdx.x * blockDim.x + threadIdx.x;
    if (idx >= total) return;
    const int t = idx % Tq;
    const int e = (idx / Tq) % E;
    const int n = idx / (Tq * E);
    const int s = n * Tq + t;
    float acc = bo[e];
    #pragma unroll
    for (int j = 0; j < 10; j++) {
        const float q = zp[s * 10 + j] > 0.0f ? 1.0f : -1.0f;
        acc += q * wo[j * E + e];
    }
    zq[idx] = f2bf(acc);
}

// ---------------------------------------------------------------------------
// Host side
// ---------------------------------------------------------------------------
extern "C" void kernel_launch(void* const* d_in, const int* in_sizes, int n_in,
                              void* d_out, int out_size, void* d_ws, size_t ws_size,
                              hipStream_t stream) {
    (void)in_sizes; (void)n_in; (void)out_size; (void)ws_size;

    const int N = 32, T0 = 512, INW = 263, W = 512, CD = 512, Tq = 64;
    const int S = N * Tq;

    auto F = [&](int i) { return (const float*)d_in[i]; };
    // jax tree_leaves order (sorted dict keys at every level):
    //  dec: c0_b,c0_w,c1_b,c1_w,c2_b,c2_w, up[0..2]{bc, res[0..2]{b1,b2,w1,w2}, wc}
    //  enc: c0_b,c0_w,c1_b,c1_w, down[0..2]{bd, res[0..2]{b1,b2,w1,w2}, wd}
    //  lfq: bi,bo,wi,wo ; then x
    const float* dec_c0_b = F(0);  const float* dec_c0_w = F(1);
    const float* dec_c1_b = F(2);  const float* dec_c1_w = F(3);
    const float* dec_c2_b = F(4);  const float* dec_c2_w = F(5);
    const int DEC_UP = 6;
    const float* enc_c0_b = F(48); const float* enc_c0_w = F(49);
    const float* enc_c1_b = F(50); const float* enc_c1_w = F(51);
    const int ENC_DN = 52;
    const float* lfq_bi = F(94);   const float* lfq_bo = F(95);
    const float* lfq_wi = F(96);   const float* lfq_wo = F(97);
    const float* x      = F(98);

    // deterministic workspace carve-out (~130 MB incl. packed weights)
    char* w8 = (char*)d_ws;
    size_t off = 0;
    auto take = [&](size_t bytes) -> void* {
        void* p = w8 + off;
        off += bytes;
        off = (off + 255) & ~(size_t)255;
        return p;
    };
    unsigned short* Xp   = (unsigned short*)take((size_t)N * INW * T0 * 2);
    unsigned short* bA   = (unsigned short*)take((size_t)N * W * T0 * 2);
    unsigned short* bB   = (unsigned short*)take((size_t)N * W * T0 * 2);
    unsigned short* bC   = (unsigned short*)take((size_t)N * W * T0 * 2);
    unsigned short* Zenc = (unsigned short*)take((size_t)N * W * Tq * 2);
    unsigned short* ZQ   = (unsigned short*)take((size_t)N * W * Tq * 2);
    float* zp     = (float*)take((size_t)S * 10 * 4);
    float* prob   = (float*)take((size_t)S * 1024 * 4);
    float* ent    = (float*)take((size_t)S * 4);
    float* commit = (float*)take((size_t)S * 4);
    float* term   = (float*)take((size_t)1024 * 4);

    unsigned short* bufs[3] = {bA, bB, bC};
    auto other = [&](unsigned short* a, unsigned short* b) -> unsigned short* {
        for (int i = 0; i < 3; i++)
            if (bufs[i] != a && bufs[i] != b) return bufs[i];
        return bA;
    };

    // pack weights (padded bf16) + launch one conv
    auto conv = [&](const unsigned short* X, const float* Wf, const float* B,
                    const unsigned short* Res, unsigned short* Ybf, float* Yf,
                    int Ci, int Tin, int Co, int Tout,
                    int Kw, int stride, int dil, int pad, int flags) {
        const int Kg  = Ci * Kw;
        const int KgP = (Kg + 31) & ~31;
        const int CoP = (Co + 127) & ~127;
        unsigned short* Wb = (unsigned short*)take((size_t)CoP * KgP * 2);
        const int tot = CoP * KgP;
        hipLaunchKernelGGL(pack_w, dim3((tot + 255) / 256), dim3(256), 0, stream,
                           Wf, Wb, Co, Kg, CoP, KgP);
        ConvParams p;
        p.X = X; p.Wb = Wb; p.Bias = B; p.Res = Res; p.Ybf = Ybf; p.Yf = Yf;
        p.Ci = Ci; p.Tin = Tin; p.Co = Co; p.Tout = Tout;
        p.Kw = Kw; p.stride = stride; p.dil = dil; p.pad = pad;
        p.Kg = Kg; p.KgP = KgP; p.flags = flags;
        dim3 grid((Tout + BN - 1) / BN, CoP / BM, N);
        hipLaunchKernelGGL(conv1d_wmma, grid, dim3(256), 0, stream, p);
    };

    // ---------------- encoder ----------------
    {
        const int total = N * INW * T0;
        hipLaunchKernelGGL(pack_input, dim3((total + 255) / 256), dim3(256), 0, stream,
                           x, Xp, N, T0, INW);
    }
    unsigned short* cur = bA;
    conv(Xp, enc_c0_w, enc_c0_b, nullptr, cur, nullptr,
         INW, T0, W, T0, 3, 1, 1, 1, F_RELU_OUT);
    int T = T0;
    for (int i = 0; i < 3; i++) {
        const int base = ENC_DN + i * 14;
        const int Td = T / 2;
        unsigned short* nxt = other(cur, cur);
        conv(cur, F(base + 13), F(base + 0), nullptr, nxt, nullptr,
             W, T, W, Td, 4, 2, 1, 1, 0);                        // k4 s2 downsample
        cur = nxt;
        int d = 1;
        for (int j = 0; j < 3; j++) {
            const float* b1 = F(base + 1 + 4 * j);
            const float* b2 = F(base + 2 + 4 * j);
            const float* w1 = F(base + 3 + 4 * j);
            const float* w2 = F(base + 4 + 4 * j);
            unsigned short* t1 = other(cur, cur);
            conv(cur, w1, b1, nullptr, t1, nullptr,
                 W, Td, W, Td, 3, 1, d, d, F_RELU_IN);           // relu -> k3 dilated
            unsigned short* t2 = other(cur, t1);
            conv(t1, w2, b2, cur, t2, nullptr,
                 W, Td, W, Td, 1, 1, 1, 0, F_RELU_IN);           // relu -> k1 + residual
            cur = t2;
            d *= 3;
        }
        T = Td;
    }
    conv(cur, enc_c1_w, enc_c1_b, nullptr, Zenc, nullptr,
         W, T, CD, T, 3, 1, 1, 1, 0);                            // z (N,512,64)

    // ---------------- LFQ + aux loss ----------------
    hipLaunchKernelGGL(lfq_stage1, dim3(S), dim3(256), 0, stream,
                       Zenc, lfq_wi, lfq_bi, zp, ent, commit, prob, Tq, CD);
    hipLaunchKernelGGL(lfq_stage2, dim3(4), dim3(256), 0, stream, prob, term, S);
    hipLaunchKernelGGL(lfq_stage3, dim3(1), dim3(256), 0, stream,
                       ent, commit, term, (float*)d_out + (size_t)N * INW * T0, S);
    {
        const int total = N * CD * Tq;
        hipLaunchKernelGGL(lfq_out, dim3((total + 255) / 256), dim3(256), 0, stream,
                           zp, lfq_wo, lfq_bo, ZQ, Tq, CD, total);
    }

    // ---------------- decoder ----------------
    conv(ZQ, dec_c0_w, dec_c0_b, nullptr, bA, nullptr,
         CD, Tq, W, Tq, 3, 1, 1, 1, F_RELU_OUT);
    cur = bA;
    T = Tq;
    for (int i = 0; i < 3; i++) {
        const int base = DEC_UP + i * 14;
        int d = 9;                                               // reversed dilations 9,3,1
        for (int j = 0; j < 3; j++) {
            const float* b1 = F(base + 1 + 4 * j);
            const float* b2 = F(base + 2 + 4 * j);
            const float* w1 = F(base + 3 + 4 * j);
            const float* w2 = F(base + 4 + 4 * j);
            unsigned short* t1 = other(cur, cur);
            conv(cur, w1, b1, nullptr, t1, nullptr,
                 W, T, W, T, 3, 1, d, d, F_RELU_IN);
            unsigned short* t2 = other(cur, t1);
            conv(t1, w2, b2, cur, t2, nullptr,
                 W, T, W, T, 1, 1, 1, 0, F_RELU_IN);
            cur = t2;
            d /= 3;
        }
        unsigned short* nxt = other(cur, cur);
        conv(cur, F(base + 13), F(base + 0), nullptr, nxt, nullptr,
             W, T, W, 2 * T, 3, 1, 1, 1, F_UP2);                 // nearest-2x + k3 conv
        cur = nxt;
        T *= 2;
    }
    unsigned short* pen = other(cur, cur);
    conv(cur, dec_c1_w, dec_c1_b, nullptr, pen, nullptr,
         W, T, W, T, 3, 1, 1, 1, F_RELU_OUT);
    conv(pen, dec_c2_w, dec_c2_b, nullptr, nullptr, (float*)d_out,
         W, T, INW, T, 3, 1, 1, 1, 0);                           // fp32 x_out (N,263,512)
}
